// GPR_GCN_3582002725351
// MI455X (gfx1250) — matrix operand order repaired and checked by reading, not compiled, not run on hardware
//
#include <hip/hip_runtime.h>
#include <hip/hip_bf16.h>
#include <cstdint>
#include <cstddef>

// ---------------------------------------------------------------------------
// MI455X / gfx1250: wave32, WMMA 16x16x32 f16 -> f32
// ---------------------------------------------------------------------------
typedef __attribute__((ext_vector_type(16))) _Float16 v16h;
typedef __attribute__((ext_vector_type(8)))  _Float16 v8h;
typedef __attribute__((ext_vector_type(8)))  float    v8f;
typedef __attribute__((ext_vector_type(4)))  float    v4f;

#define NN   512            // nodes
#define TT   64             // seq len
#define PPAT (NN * TT)      // 32768 patches

__device__ inline v8f zero8() {
  v8f z;
#pragma unroll
  for (int i = 0; i < 8; ++i) z[i] = 0.f;
  return z;
}
__device__ inline v8h zero8h() {
  v8h z;
#pragma unroll
  for (int i = 0; i < 8; ++i) z[i] = (_Float16)0.f;
  return z;
}
__device__ inline v16h combine16(v8h lo, v8h hi) {
  return __builtin_shufflevector(lo, hi, 0, 1, 2, 3, 4, 5, 6, 7,
                                 8, 9, 10, 11, 12, 13, 14, 15);
}

__device__ inline v8f wmma16(v16h a, v16h b, v8f c) {
  return __builtin_amdgcn_wmma_f32_16x16x32_f16(false, a, false, b,
                                                (short)0, c, false, false);
}

// gfx1250 has v_tanh_f32; use it when clang exposes the builtin.
#if __has_builtin(__builtin_amdgcn_tanhf)
__device__ inline float fast_tanh(float x) { return __builtin_amdgcn_tanhf(x); }
#else
__device__ inline float fast_tanh(float x) { return tanhf(x); }
#endif
__device__ inline float sigm(float x) {
  return __builtin_amdgcn_rcpf(1.f + __expf(-x));
}

// Device-scope generation barrier (grid = 16 blocks, all co-resident).
__device__ inline void grid_barrier16(unsigned* cnt, unsigned* gen) {
  __syncthreads();
  if (threadIdx.x == 0) {
    __threadfence();                                   // release prior writes
    unsigned g = __hip_atomic_load(gen, __ATOMIC_RELAXED, __HIP_MEMORY_SCOPE_AGENT);
    unsigned a = __hip_atomic_fetch_add(cnt, 1u, __ATOMIC_ACQ_REL, __HIP_MEMORY_SCOPE_AGENT);
    if (a == 15u) {
      __hip_atomic_store(cnt, 0u, __ATOMIC_RELAXED, __HIP_MEMORY_SCOPE_AGENT);
      __hip_atomic_fetch_add(gen, 1u, __ATOMIC_ACQ_REL, __HIP_MEMORY_SCOPE_AGENT);
    } else {
      while (__hip_atomic_load(gen, __ATOMIC_ACQUIRE, __HIP_MEMORY_SCOPE_AGENT) == g) {}
    }
  }
  __syncthreads();
}

// B fragment: Bt stored [N, K] row-major (W^T, K mult of 32).
__device__ inline v16h load_b_frag(const _Float16* __restrict__ Bt, int ldk,
                                   int col_base, int k0, int lane) {
  int n  = col_base + (lane & 15);
  int ko = k0 + ((lane >> 4) << 3);
  const _Float16* p = Bt + (size_t)n * ldk + ko;
  return combine16(*(const v8h*)p, *(const v8h*)(p + 16));   // 2x b128
}

// A fragment from contiguous f16 row
__device__ inline v16h load_a16(const _Float16* __restrict__ pA, int k0, int lane) {
  int ko = k0 + ((lane >> 4) << 3);
  v8h lo = *(const v8h*)(pA + ko);
  v8h hi = *(const v8h*)(pA + ko + 16);
  return combine16(lo, hi);
}

// A fragment from contiguous f32 row (converted to f16)
__device__ inline v8h cvt_chunk(const float* __restrict__ p) {
  v4f a = *(const v4f*)p;
  v4f b = *(const v4f*)(p + 4);
  v8h r;
#pragma unroll
  for (int e = 0; e < 4; ++e) {
    r[e]     = (_Float16)a[e];
    r[4 + e] = (_Float16)b[e];
  }
  return r;
}
__device__ inline v16h load_a32(const float* __restrict__ pA, int k0, int lane) {
  int ko = k0 + ((lane >> 4) << 3);
  return combine16(cvt_chunk(pA + ko), cvt_chunk(pA + ko + 16));
}

template <typename AT>
__device__ inline v16h load_a(const AT* pA, int k0, int lane);
template <>
__device__ inline v16h load_a<float>(const float* pA, int k0, int lane) {
  return load_a32(pA, k0, lane);
}
template <>
__device__ inline v16h load_a<_Float16>(const _Float16* pA, int k0, int lane) {
  return load_a16(pA, k0, lane);
}

// ---------------------------------------------------------------------------
// Generic WMMA GEMM: NT n-tiles per wave; B loads batched ahead of the WMMAs.
// epilogue: (+bias) -> relu -> (+addend) -> store f32 C and/or transposed f16.
// Requires: M%16==0, N%(16*NT)==0, K%32==0, A fully materialized [M,K].
// ---------------------------------------------------------------------------
template <typename AT, int NT>
__global__ void k_gemm(const AT* __restrict__ A, int K,
                       const _Float16* __restrict__ Bt,
                       const float* __restrict__ bias,
                       const float* __restrict__ addend, int relu,
                       float* __restrict__ Cf, _Float16* __restrict__ CtT,
                       int M, int N) {
  int tid  = blockIdx.x * 256 + threadIdx.x;
  int wave = tid >> 5, lane = tid & 31;
  int ngrp  = N / (16 * NT);
  int tiles = (M >> 4) * ngrp;
  if (wave >= tiles) return;                 // wave-uniform: EXEC stays all-1
  int mt = wave / ngrp, ng = wave % ngrp;

  const AT* pA = A + (size_t)(mt * 16 + (lane & 15)) * K;
  v8f acc[NT];
#pragma unroll
  for (int j = 0; j < NT; ++j) acc[j] = zero8();

  for (int k0 = 0; k0 < K; k0 += 32) {
    __builtin_prefetch(pA + k0 + 32, 0, 1);  // global_prefetch_b8
    v16h a = load_a<AT>(pA, k0, lane);
    v16h b[NT];
#pragma unroll
    for (int j = 0; j < NT; ++j)
      b[j] = load_b_frag(Bt, K, (ng * NT + j) * 16, k0, lane);
#pragma unroll
    for (int j = 0; j < NT; ++j) acc[j] = wmma16(a, b[j], acc[j]);
  }
  int mbase = mt * 16 + ((lane >> 4) << 3);
#pragma unroll
  for (int j = 0; j < NT; ++j) {
    int n = (ng * NT + j) * 16 + (lane & 15);
    float bn = bias ? bias[n] : 0.f;
#pragma unroll
    for (int v = 0; v < 8; ++v) {
      int m = mbase + v;
      float val = acc[j][v] + bn;
      if (relu) val = fmaxf(val, 0.f);
      if (addend) val += addend[(size_t)m * N + n];
      if (Cf)  Cf[(size_t)m * N + n]  = val;
      if (CtT) CtT[(size_t)n * M + m] = (_Float16)val;  // B-ready transposed f16
    }
  }
}

// ---------------------------------------------------------------------------
// conv3x3 SAME as WMMA GEMM, fused im2col + fused 2x2 maxpool, NHWC activations.
// K-dim is tap-major: k = tap*Cin + cin -> 8-wide chunks are single-tap,
// contiguous-in-cin => one b128 load + one bounds test per chunk.
// Row order r = ((p*H2+y2)*W2+x2)*4 + quadrant => pool groups = 4 rows.
// ---------------------------------------------------------------------------
template <int Cin, int H, int W>
__device__ inline v8h conv_chunk(const _Float16* __restrict__ In,
                                 int p, int y, int x, int kb) {
  constexpr int LG = (Cin == 32) ? 5 : 6;
  int tap  = kb >> LG;
  int cin0 = kb & (Cin - 1);
  int dy = tap / 3, dx = tap - 3 * dy;
  int yy = y + dy - 1, xx = x + dx - 1;
  v8h c = zero8h();
  if (yy >= 0 && yy < H && xx >= 0 && xx < W)
    c = *(const v8h*)(In + ((size_t)(p * H + yy) * W + xx) * Cin + cin0);
  return c;
}

template <int Cin, int H, int W, int Cout>
__global__ void k_conv_gemm_pool(const _Float16* __restrict__ In,
                                 const _Float16* __restrict__ Bt,
                                 const float* __restrict__ bias,
                                 _Float16* __restrict__ Pooled, int P) {
  constexpr int K  = Cin * 9;                 // 288 / 576 (mult of 32)
  constexpr int NT = 4;
  constexpr int H2 = H / 2, W2 = W / 2;
  int tid  = blockIdx.x * 256 + threadIdx.x;
  int wave = tid >> 5, lane = tid & 31;
  int ngrp  = Cout / (16 * NT);
  int M     = P * H * W;
  int tiles = (M >> 4) * ngrp;
  if (wave >= tiles) return;
  int mt = wave / ngrp, ng = wave % ngrp;

  int r = mt * 16 + (lane & 15);
  int q = r & 3, sp = r >> 2;
  int x2 = sp % W2; int t = sp / W2;
  int y2 = t % H2;  int p = t / H2;
  int y = 2 * y2 + (q >> 1), x = 2 * x2 + (q & 1);

  v8f acc[NT];
#pragma unroll
  for (int j = 0; j < NT; ++j) acc[j] = zero8();

  for (int k0 = 0; k0 < K; k0 += 32) {
    int ko = k0 + ((lane >> 4) << 3);
    v16h a = combine16(conv_chunk<Cin, H, W>(In, p, y, x, ko),
                       conv_chunk<Cin, H, W>(In, p, y, x, ko + 16));
    v16h b[NT];
#pragma unroll
    for (int j = 0; j < NT; ++j)
      b[j] = load_b_frag(Bt, K, (ng * NT + j) * 16, k0, lane);
#pragma unroll
    for (int j = 0; j < NT; ++j) acc[j] = wmma16(a, b[j], acc[j]);
  }
  int pr0 = mt * 4 + ((lane >> 4) << 1);
#pragma unroll
  for (int j = 0; j < NT; ++j) {
    int c = (ng * NT + j) * 16 + (lane & 15);
    float bn = bias[c];
    float m0 = fmaxf(fmaxf(acc[j][0], acc[j][1]), fmaxf(acc[j][2], acc[j][3])) + bn;
    float m1 = fmaxf(fmaxf(acc[j][4], acc[j][5]), fmaxf(acc[j][6], acc[j][7])) + bn;
#pragma unroll
    for (int jj = 0; jj < 2; ++jj) {
      int pr = pr0 + jj;            // pooled row = (p*H2+y2)*W2+x2
      Pooled[(size_t)pr * Cout + c] = (_Float16)(jj ? m1 : m0);
    }
  }
}

// ---------------------------------------------------------------------------
// conv1 (Cin=1, K=9 -- WMMA-hostile) + pool, direct VALU. Out [P,4,4,32] NHWC.
// ---------------------------------------------------------------------------
__global__ void k_conv1pool(const float* __restrict__ In,
                            const float* __restrict__ W0,
                            const float* __restrict__ b0,
                            _Float16* __restrict__ Out, int P) {
  int idx = blockIdx.x * 256 + threadIdx.x;   // ((p*16 + s)*32 + c), NHWC
  if (idx >= P * 16 * 32) return;
  int c = idx & 31, s = (idx >> 5) & 15, p = idx >> 9;
  int x2 = s & 3, y2 = s >> 2;
  const float* ip = In + (size_t)p * 64;
  float best = -1e30f;
#pragma unroll
  for (int sy = 0; sy < 2; ++sy)
#pragma unroll
    for (int sx = 0; sx < 2; ++sx) {
      int y = 2 * y2 + sy, x = 2 * x2 + sx;
      float acc = b0[c];
#pragma unroll
      for (int dy = 0; dy < 3; ++dy)
#pragma unroll
        for (int dx = 0; dx < 3; ++dx) {
          int yy = y + dy - 1, xx = x + dx - 1;
          if (yy >= 0 && yy < 8 && xx >= 0 && xx < 8)
            acc += ip[yy * 8 + xx] * W0[c * 9 + dy * 3 + dx];
        }
      best = fmaxf(best, acc);
    }
  Out[idx] = (_Float16)best;
}

// ---------------------------------------------------------------------------
// Persistent bidirectional-LSTM direction kernel.
// Grid = 16 blocks x 256 (128 waves = full tile set: mt 0..31, gate-col nt 0..3).
// Each wave owns one (mt,nt) tile for ALL timesteps of all 4 layers:
//   - cell state stays in registers (v8f per lane) for the whole sequence
//   - gate weights get hoisted into high VGPRs (v256+) across the time loop
//   - X-part and H-part K-slices are split at compile time (Cx % 32 == 0), so
//     all A loads are pure global_load_b128 (no FLAT, no pointer selects);
//     step 0 skips the H-part entirely (h(-1) == 0)
//   - device-scope generation barrier between steps (h(t) -> A(t+1))
// ---------------------------------------------------------------------------
template <int Cx>
__device__ void lstm_layer(const _Float16* __restrict__ Xseq, int reverse,
                           const _Float16* __restrict__ Bt,   // [256, Cx+64]
                           const float* __restrict__ bias,    // [256]
                           _Float16* __restrict__ Yout,       // [T,512,64]
                           int mt, int nt, int lane,
                           unsigned* cnt, unsigned* gen) {
  constexpr int K = Cx + 64;
  int r     = mt * 16 + (lane & 15);
  int nwg   = nt * 16 + (lane & 15);
  int mbase = mt * 16 + ((lane >> 4) << 3);
  float bi = bias[nwg], bf = bias[64 + nwg], bo = bias[128 + nwg], bg = bias[192 + nwg];
  v8f cstate = zero8();                        // cell state lives in registers

  for (int s = 0; s < TT; ++s) {
    const _Float16* Xrow =
        Xseq + (size_t)(reverse ? (TT - 1 - s) : s) * NN * Cx + (size_t)r * Cx;
    v8f acc[4] = {zero8(), zero8(), zero8(), zero8()};

    // X part: K-slices [0, Cx)
#pragma unroll
    for (int k0 = 0; k0 < Cx; k0 += 32) {
      v16h a = load_a16(Xrow, k0, lane);
      v16h b[4];
#pragma unroll
      for (int g = 0; g < 4; ++g)
        b[g] = load_b_frag(Bt, K, g * 64 + nt * 16, k0, lane);
#pragma unroll
      for (int g = 0; g < 4; ++g) acc[g] = wmma16(a, b[g], acc[g]);
    }
    // H part: K-slices [Cx, Cx+64); h(-1)==0 -> skip at s==0 (uniform branch)
    if (s > 0) {
      const _Float16* Hrow = Yout + (size_t)(s - 1) * NN * 64 + (size_t)r * 64;
#pragma unroll
      for (int k0 = 0; k0 < 64; k0 += 32) {
        v16h a = load_a16(Hrow, k0, lane);
        v16h b[4];
#pragma unroll
        for (int g = 0; g < 4; ++g)
          b[g] = load_b_frag(Bt, K, g * 64 + nt * 16, Cx + k0, lane);
#pragma unroll
        for (int g = 0; g < 4; ++g) acc[g] = wmma16(a, b[g], acc[g]);
      }
    }

    _Float16* Ho = Yout + (size_t)s * NN * 64;
#pragma unroll
    for (int v = 0; v < 8; ++v) {
      int m = mbase + v;
      float iv = acc[0][v] + bi;
      float fv = acc[1][v] + bf;
      float ov = acc[2][v] + bo;
      float gv = acc[3][v] + bg;
      float cnew = sigm(fv) * cstate[v] + sigm(iv) * fast_tanh(gv);
      cstate[v] = cnew;
      Ho[(size_t)m * 64 + nwg] = (_Float16)(sigm(ov) * fast_tanh(cnew));
    }
    grid_barrier16(cnt, gen);                  // h(s) visible before step s+1
  }
}

__global__ void __launch_bounds__(256, 1)
k_lstm_dir(const _Float16* __restrict__ X0,     // [T,512,128]
           const _Float16* __restrict__ Bt0, const _Float16* __restrict__ Bt1,
           const _Float16* __restrict__ Bt2, const _Float16* __restrict__ Bt3,
           const float* __restrict__ b0, const float* __restrict__ b1,
           const float* __restrict__ b2, const float* __restrict__ b3,
           _Float16* __restrict__ Ya, _Float16* __restrict__ Yb,
           unsigned* cnt, unsigned* gen, int reverse) {
  int tid  = blockIdx.x * 256 + threadIdx.x;
  int wave = tid >> 5, lane = tid & 31;
  int mt = wave >> 2, nt = wave & 3;           // grid is exactly 128 waves
  lstm_layer<128>(X0, reverse, Bt0, b0, Ya, mt, nt, lane, cnt, gen);
  lstm_layer<64>(Ya, 0, Bt1, b1, Yb, mt, nt, lane, cnt, gen);
  lstm_layer<64>(Yb, 0, Bt2, b2, Ya, mt, nt, lane, cnt, gen);
  lstm_layer<64>(Ya, 0, Bt3, b3, Yb, mt, nt, lane, cnt, gen);
}

// ---------------------------------------------------------------------------
// Small prep / glue kernels
// ---------------------------------------------------------------------------
__global__ void k_prep_linear_t(const float* __restrict__ W, _Float16* __restrict__ Bt,
                                int K, int N, int Kpad) {   // W [K,N] -> Bt [N,Kpad]
  int idx = blockIdx.x * 256 + threadIdx.x;
  if (idx >= N * Kpad) return;
  int n = idx / Kpad, k = idx % Kpad;
  Bt[idx] = (k < K) ? (_Float16)W[(size_t)k * N + n] : (_Float16)0.f;
}

template <int Cin>
__global__ void k_prep_conv_t(const float* __restrict__ W, _Float16* __restrict__ Bt,
                              int Cout) {  // [Cout,Cin,3,3] -> Bt[n][tap*Cin+cin]
  constexpr int K = Cin * 9;
  int idx = blockIdx.x * 256 + threadIdx.x;
  if (idx >= Cout * K) return;
  int n = idx / K, k = idx % K;
  int tap = k >> ((Cin == 32) ? 5 : 6);
  int cin = k & (Cin - 1);
  Bt[idx] = (_Float16)W[((size_t)n * Cin + cin) * 9 + tap];
}

__global__ void k_prep_lstm_t(const float* __restrict__ W, _Float16* __restrict__ Bt,
                              int K) {                      // center tap of [256,K,3,3]
  int idx = blockIdx.x * 256 + threadIdx.x;
  if (idx >= 256 * K) return;
  Bt[idx] = (_Float16)W[(size_t)idx * 9 + 4];
}

__global__ void k_pad_pos(const float* __restrict__ In, float* __restrict__ Out) {
  int idx = blockIdx.x * 256 + threadIdx.x;   // [512,16] -> [512,32] zero-padded
  if (idx >= NN * 32) return;
  int m = idx >> 5, k = idx & 31;
  Out[idx] = (k < 16) ? In[m * 16 + k] : 0.f;
}

__global__ void k_colsum_abs(const float* __restrict__ A, float* __restrict__ cs) {
  int j = blockIdx.x * 256 + threadIdx.x;
  if (j >= NN) return;
  float s = 0.f;
  for (int i = 0; i < NN; ++i) s += fabsf(A[(size_t)i * NN + j]);
  cs[j] = fmaxf(s, 1e-12f);
}

__global__ void k_norm_A(const float* __restrict__ A, const float* __restrict__ cs,
                         _Float16* __restrict__ An) {
  int idx = blockIdx.x * 256 + threadIdx.x;
  if (idx >= NN * NN) return;
  An[idx] = (_Float16)(A[idx] / cs[idx & (NN - 1)]);
}

__global__ void k_zero_u32(unsigned* p, int n) {
  int i = blockIdx.x * 256 + threadIdx.x;
  if (i < n) p[i] = 0u;
}

__global__ void k_time_major(const _Float16* __restrict__ pooled3,
                             _Float16* __restrict__ X0) {   // [P,128] -> [T,N,128]
  int idx = blockIdx.x * 256 + threadIdx.x;
  if (idx >= TT * NN * 128) return;
  int c = idx & 127, n = (idx >> 7) & (NN - 1), t = idx >> 16;
  X0[idx] = pooled3[((size_t)(n << 6) + t) * 128 + c];
}

__global__ void k_copy_cond(const _Float16* __restrict__ Yt,
                            _Float16* __restrict__ cond, int off) {
  int idx = blockIdx.x * 256 + threadIdx.x;
  if (idx >= NN * 64) return;
  int m = idx >> 6, j = idx & 63;
  cond[(size_t)m * 128 + off + j] = Yt[idx];
}

__global__ void k_build_fused(const float* __restrict__ pos,
                              const _Float16* __restrict__ cond,
                              _Float16* __restrict__ fused) {
  int idx = blockIdx.x * 256 + threadIdx.x;
  if (idx >= NN * 192) return;
  int m = idx / 192, k = idx % 192;
  fused[idx] = (k < 64) ? (_Float16)pos[(size_t)m * 64 + k]
                        : cond[(size_t)m * 128 + (k - 64)];
}

// ---------------------------------------------------------------------------
// Host side
// ---------------------------------------------------------------------------
static inline int cdiv(long long a, int b) { return (int)((a + b - 1) / b); }

enum {
  IN_POS = 0, IN_IMG, IN_SPLIT, IN_NNODE, IN_A,
  P_CLS_W, P_CLS_B,
  P_G1A_W, P_G1A_B, P_G1U_W, P_G1U_B, P_G1W_W, P_G1W_B,
  P_G2A_W, P_G2A_B, P_G2U_W, P_G2U_B, P_G2W_W, P_G2W_B,
  P_LF0_W, P_LF0_B, P_LF1_W, P_LF1_B, P_LF2_W, P_LF2_B, P_LF3_W, P_LF3_B,
  P_LR0_W, P_LR0_B, P_LR1_W, P_LR1_B, P_LR2_W, P_LR2_B, P_LR3_W, P_LR3_B,
  P_PIX0_W, P_PIX0_B, P_PIX1_W, P_PIX1_B, P_PIX2_W, P_PIX2_B,
  P_POS0_W, P_POS0_B, P_POS1_W, P_POS1_B, P_POS2_W, P_POS2_B, P_POS3_W, P_POS3_B
};

struct Bump {
  char* base; size_t off;
  void* take(size_t bytes) {
    off = (off + 255) & ~(size_t)255;
    void* p = base + off;
    off += bytes;
    return p;
  }
};

extern "C" void kernel_launch(void* const* d_in, const int* in_sizes, int n_in,
                              void* d_out, int out_size, void* d_ws, size_t ws_size,
                              hipStream_t stream) {
  (void)in_sizes; (void)n_in; (void)out_size; (void)ws_size;
  auto F = [&](int i) { return (const float*)d_in[i]; };

  Bump ws{(char*)d_ws, 0};
  auto h16 = [&](size_t n) { return (_Float16*)ws.take(n * sizeof(_Float16)); };
  auto f32 = [&](size_t n) { return (float*)ws.take(n * sizeof(float)); };

  // ---- weight buffers (f16, B-transposed, K mult of 32) ----
  _Float16* pos_t[4] = {h16(512 * 32), h16(256 * 512), h16(128 * 256), h16(64 * 128)};
  _Float16* conv2t = h16(64 * 288);
  _Float16* conv3t = h16(128 * 576);
  _Float16* lstm_t[2][4];
  int lstmK[4] = {192, 128, 128, 128};
  for (int d = 0; d < 2; ++d)
    for (int l = 0; l < 4; ++l) lstm_t[d][l] = h16(256 * lstmK[l]);
  _Float16* ga_t[2] = {h16(256 * 192), h16(256 * 192)};
  _Float16* gu_t[2] = {h16(256 * 192), h16(256 * 192)};
  _Float16* gw_t[2] = {h16(64 * 256), h16(64 * 256)};
  _Float16* cls_t   = h16(16 * 64);
  _Float16* An      = h16((size_t)NN * NN);
  float*    colsum  = f32(NN);
  unsigned* barrier = (unsigned*)ws.take(2 * sizeof(unsigned));  // {cnt, gen}

  // ---- activations ----
  float* posPad = f32((size_t)NN * 32);
  float* x1   = f32((size_t)NN * 512);
  float* x2   = f32((size_t)NN * 256);
  float* x3   = f32((size_t)NN * 128);
  float* posA = f32((size_t)NN * 64);
  float* posB = f32((size_t)NN * 64);
  _Float16* pooled1 = h16((size_t)PPAT * 16 * 32);  // [P,4,4,32] NHWC
  _Float16* pooled2 = h16((size_t)PPAT * 4 * 64);   // [P,2,2,64] NHWC
  _Float16* pooled3 = h16((size_t)PPAT * 128);      // [P,128]
  _Float16* X0 = h16((size_t)TT * NN * 128);
  _Float16* Ya = h16((size_t)TT * NN * 64);
  _Float16* Yb = h16((size_t)TT * NN * 64);
  _Float16* cond  = h16((size_t)NN * 128);
  _Float16* fused = h16((size_t)NN * 192);
  _Float16* t1T   = h16((size_t)256 * NN);
  float*    t2    = f32((size_t)NN * 256);
  float*    gbuf  = f32((size_t)NN * 256);

  // ---- weight prep ----
  auto prep_lin = [&](int wi, _Float16* bt, int K, int N, int Kpad) {
    k_prep_linear_t<<<cdiv((long long)N * Kpad, 256), 256, 0, stream>>>(F(wi), bt, K, N, Kpad);
  };
  prep_lin(P_POS0_W, pos_t[0], 16, 512, 32);
  prep_lin(P_POS1_W, pos_t[1], 512, 256, 512);
  prep_lin(P_POS2_W, pos_t[2], 256, 128, 256);
  prep_lin(P_POS3_W, pos_t[3], 128, 64, 128);
  prep_lin(P_G1A_W, ga_t[0], 192, 256, 192);
  prep_lin(P_G1U_W, gu_t[0], 192, 256, 192);
  prep_lin(P_G1W_W, gw_t[0], 256, 64, 256);
  prep_lin(P_G2A_W, ga_t[1], 192, 256, 192);
  prep_lin(P_G2U_W, gu_t[1], 192, 256, 192);
  prep_lin(P_G2W_W, gw_t[1], 256, 64, 256);
  prep_lin(P_CLS_W, cls_t, 64, 16, 64);
  k_prep_conv_t<32><<<cdiv(64 * 288, 256), 256, 0, stream>>>(F(P_PIX1_W), conv2t, 64);
  k_prep_conv_t<64><<<cdiv(128 * 576, 256), 256, 0, stream>>>(F(P_PIX2_W), conv3t, 128);
  int lstmWidx[2][4] = {{P_LF0_W, P_LF1_W, P_LF2_W, P_LF3_W},
                        {P_LR0_W, P_LR1_W, P_LR2_W, P_LR3_W}};
  int lstmBidx[2][4] = {{P_LF0_B, P_LF1_B, P_LF2_B, P_LF3_B},
                        {P_LR0_B, P_LR1_B, P_LR2_B, P_LR3_B}};
  for (int d = 0; d < 2; ++d)
    for (int l = 0; l < 4; ++l)
      k_prep_lstm_t<<<cdiv(256 * lstmK[l], 256), 256, 0, stream>>>(
          F(lstmWidx[d][l]), lstm_t[d][l], lstmK[l]);

  // ---- An = column-L1-normalized A (f16) ----
  k_colsum_abs<<<cdiv(NN, 256), 256, 0, stream>>>(F(IN_A), colsum);
  k_norm_A<<<cdiv((long long)NN * NN, 256), 256, 0, stream>>>(F(IN_A), colsum, An);

  // ---- GEMM launchers (NT=4 when N%64==0, else NT=1) ----
  auto gemm32 = [&](const float* A, int K, const _Float16* Bt,
                    const float* bias, const float* add, int relu,
                    float* Cf, _Float16* CtT, int M, int N) {
    if (N % 64 == 0) {
      int waves = (M / 16) * (N / 64);
      k_gemm<float, 4><<<cdiv((long long)waves * 32, 256), 256, 0, stream>>>(
          A, K, Bt, bias, add, relu, Cf, CtT, M, N);
    } else {
      int waves = (M / 16) * (N / 16);
      k_gemm<float, 1><<<cdiv((long long)waves * 32, 256), 256, 0, stream>>>(
          A, K, Bt, bias, add, relu, Cf, CtT, M, N);
    }
  };
  auto gemm16 = [&](const _Float16* A, int K, const _Float16* Bt,
                    const float* bias, const float* add, int relu,
                    float* Cf, _Float16* CtT, int M, int N) {
    int waves = (M / 16) * (N / 64);
    k_gemm<_Float16, 4><<<cdiv((long long)waves * 32, 256), 256, 0, stream>>>(
        A, K, Bt, bias, add, relu, Cf, CtT, M, N);
  };

  // ---- pos MLP: 4 x (linear + relu) ----
  k_pad_pos<<<cdiv(NN * 32, 256), 256, 0, stream>>>(F(IN_POS), posPad);
  gemm32(posPad, 32, pos_t[0], F(P_POS0_B), nullptr, 1, x1, nullptr, NN, 512);
  gemm32(x1, 512, pos_t[1], F(P_POS1_B), nullptr, 1, x2, nullptr, NN, 256);
  gemm32(x2, 256, pos_t[2], F(P_POS2_B), nullptr, 1, x3, nullptr, NN, 128);
  gemm32(x3, 128, pos_t[3], F(P_POS3_B), nullptr, 1, posA, nullptr, NN, 64);

  // ---- pix CNN (NHWC) ----
  k_conv1pool<<<cdiv((long long)PPAT * 16 * 32, 256), 256, 0, stream>>>(
      F(IN_IMG), F(P_PIX0_W), F(P_PIX0_B), pooled1, PPAT);
  {
    long long waves2 = ((long long)PPAT * 16 / 16) * (64 / 64);
    k_conv_gemm_pool<32, 4, 4, 64><<<cdiv(waves2 * 32, 256), 256, 0, stream>>>(
        pooled1, conv2t, F(P_PIX1_B), pooled2, PPAT);
    long long waves3 = ((long long)PPAT * 4 / 16) * (128 / 64);
    k_conv_gemm_pool<64, 2, 2, 128><<<cdiv(waves3 * 32, 256), 256, 0, stream>>>(
        pooled2, conv3t, F(P_PIX2_B), pooled3, PPAT);
  }
  k_time_major<<<cdiv((long long)TT * NN * 128, 256), 256, 0, stream>>>(pooled3, X0);

  // ---- bidirectional 4-layer LSTM: one persistent kernel per direction ----
  k_zero_u32<<<1, 256, 0, stream>>>(barrier, 2);
  for (int d = 0; d < 2; ++d) {
    k_lstm_dir<<<16, 256, 0, stream>>>(
        X0,
        lstm_t[d][0], lstm_t[d][1], lstm_t[d][2], lstm_t[d][3],
        F(lstmBidx[d][0]), F(lstmBidx[d][1]), F(lstmBidx[d][2]), F(lstmBidx[d][3]),
        Ya, Yb, barrier, barrier + 1, d);
    // layer-3 output lives in Yb; cond = [fwd h(T-1) | rev h(step 0)]
    k_copy_cond<<<cdiv(NN * 64, 256), 256, 0, stream>>>(
        Yb + (size_t)((d == 0) ? (TT - 1) : 0) * NN * 64, cond, (d == 0) ? 0 : 64);
  }

  // ---- 2x GCN: g = An @ relu(f@Wa+ba) + relu(f@Wu+bu); pos += relu(g@Ww+bw) ----
  int gaB[2] = {P_G1A_B, P_G2A_B}, guB[2] = {P_G1U_B, P_G2U_B}, gwB[2] = {P_G1W_B, P_G2W_B};
  float* posOld = posA;
  float* posNew = posB;
  for (int l = 0; l < 2; ++l) {
    k_build_fused<<<cdiv(NN * 192, 256), 256, 0, stream>>>(posOld, cond, fused);
    gemm16(fused, 192, ga_t[l], F(gaB[l]), nullptr, 1, nullptr, t1T, NN, 256);
    gemm16(fused, 192, gu_t[l], F(guB[l]), nullptr, 1, t2, nullptr, NN, 256);
    gemm16(An, NN, t1T, nullptr, t2, 0, gbuf, nullptr, NN, 256);   // An@t1 + t2
    gemm32(gbuf, 256, gw_t[l], F(gwB[l]), posOld, 1, posNew, nullptr, NN, 64);
    float* tmp = posOld; posOld = posNew; posNew = tmp;
  }

  // ---- classifier -> d_out [512,16] f32 ----
  gemm32(posOld, 64, cls_t, F(P_CLS_B), nullptr, 0, (float*)d_out, nullptr, NN, 16);
}
